// CostVolume_35038343201205
// MI455X (gfx1250) — matrix operand order repaired and checked by reading, not compiled, not run on hardware
//
#include <hip/hip_runtime.h>

typedef __attribute__((ext_vector_type(16))) __bf16 v16bf;
typedef __attribute__((ext_vector_type(2)))  __bf16 v2bf;
typedef __attribute__((ext_vector_type(8)))  float  v8f;

#define B_    4
#define C_    256
#define H_    96
#define W_    96
#define HW_   (H_ * W_)            // 9216
#define CHW_  (C_ * HW_)           // 2359296
#define P_    9                    // patch size (2*4+1)

// padded bf16x2 plane: y in [-4, 100), x in [-4, 108)
#define PH_   104
#define PW_   112                  // 448 B row stride (64B multiple)
#define PHW_  (PH_ * PW_)          // 11648 dwords per (b, c-pair) plane
#define NPADDED_ (B_ * (C_ / 2) * PHW_)   // 5,963,776 dwords per tensor

// Pack two floats into one bf16x2 dword (RNE).
__device__ __forceinline__ unsigned packbf(float lo, float hi) {
#if __has_builtin(__builtin_amdgcn_cvt_pk_bf16_f32)
    union { v2bf v; unsigned u; } p;
    p.v = __builtin_amdgcn_cvt_pk_bf16_f32(lo, hi);   // single v_cvt_pk_bf16_f32
    return p.u;
#else
    union { v2bf v; unsigned u; } p;
    p.v = (v2bf){ (__bf16)lo, (__bf16)hi };
    return p.u;
#endif
}

union Frag { unsigned u[8]; v16bf v; };

// ---- pre-pass: f32 [b,c,y,x] -> zero-haloed packed bf16x2 [b,c/2,y+4,x+4] ------
__global__ __launch_bounds__(256) void pack_pairs_padded(
    const float* __restrict__ src, unsigned* __restrict__ dst)
{
    const int i = blockIdx.x * 256 + threadIdx.x;
    if (i >= NPADDED_) return;
    const int bc2 = i / PHW_;                 // b*128 + c2
    const int r   = i - bc2 * PHW_;
    const int py  = r / PW_;
    const int px  = r - py * PW_;
    const int y   = py - 4;
    const int x   = px - 4;
    unsigned val = 0;                          // halo cells = +0.0 bf16 pair
    if (y >= 0 && y < H_ && x >= 0 && x < W_) {
        // channel 2*c2 of batch b: (b*256 + 2*c2)*HW = 2*bc2*HW
        const float* p = src + 2 * bc2 * HW_ + y * W_ + x;
        val = packbf(p[0], p[HW_]);
    }
    dst[i] = val;
}

// ---- main kernel: pure loads (halo supplies zeros), WMMA band GEMM -------------
__global__ __launch_bounds__(192) void cost_volume_wmma_padded(
    const unsigned* __restrict__ w1, const unsigned* __restrict__ w2,
    float* __restrict__ out)
{
    __shared__ float lds[6 * 512];            // 2 KB per wave: two 16x16 f32 tiles

    const int b  = blockIdx.x / H_;           // scalar per block
    const int yy = blockIdx.x % H_;

    const int tid  = threadIdx.x;
    const int lane = tid & 31;
    const int wave = tid >> 5;                // x-tile 0..5
    const int x0   = wave * 16;

    const int n    = lane & 15;
    const int half = lane >> 4;

    const unsigned* f1b = w1 + b * (C_ / 2) * PHW_;   // scalar bases
    const unsigned* f2b = w2 + b * (C_ / 2) * PHW_;
    float*       outRow = out + b * (P_ * P_) * HW_ + yy * W_;

    // A fragments: f1 padded plane at (yy+4, x0+n+4); pair pa = half*8+v
    Frag A[8];
    const int aoff = (yy + 4) * PW_ + x0 + n + 4;
    #pragma unroll
    for (int kc = 0; kc < 8; ++kc) {
        const unsigned* f1kc = f1b + kc * 16 * PHW_;
        #pragma unroll
        for (int v = 0; v < 8; ++v) {
            const int pa = (half << 3) + v;
            A[kc].u[v] = f1kc[aoff + pa * PHW_];
        }
    }

    float* myLds = lds + wave * 512;

    // tile0 B col n -> padded x = x0 + n ; tile1 -> +16 (folds into imm offset)
    int off0 = yy * PW_ + x0 + n;             // dyi=0 => padded row yy
    for (int dyi = 0; dyi < P_; ++dyi) {
        if (dyi + 1 < P_)                     // global_prefetch_b8 next row
            __builtin_prefetch(f2b + off0 + PW_, 0, 0);

        v8f c0 = {}; v8f c1 = {};
        #pragma unroll
        for (int kc = 0; kc < 8; ++kc) {
            const unsigned* f2kc = f2b + kc * 16 * PHW_;
            Frag Bf0, Bf1;
            #pragma unroll
            for (int v = 0; v < 8; ++v) {
                const int kp = (half << 3) + v;      // B pair idx = half*8 + v
                Bf0.u[v] = f2kc[off0 + kp * PHW_];
                Bf1.u[v] = f2kc[off0 + kp * PHW_ + 16];
            }
            c0 = __builtin_amdgcn_wmma_f32_16x16x32_bf16(
                     false, A[kc].v, false, Bf0.v, (short)0, c0, false, false);
            c1 = __builtin_amdgcn_wmma_f32_16x16x32_bf16(
                     false, A[kc].v, false, Bf1.v, (short)0, c1, false, false);
        }

        // transpose D tiles through LDS (lane n holds col, VGPR r -> row r+8*half)
        #pragma unroll
        for (int r = 0; r < 8; ++r) {
            myLds[((half << 3) + r) * 16 + n]       = c0[r];
            myLds[256 + ((half << 3) + r) * 16 + n] = c1[r];
        }
        __syncthreads();

        if (lane < 16) {
            const int m = lane;
            const int obase = (dyi * P_) * HW_ + x0 + m;
            #pragma unroll
            for (int dx = 0; dx < P_; ++dx) {
                const int col = m + dx;               // combined column, 0..23
                const float val = (col < 16) ? myLds[m * 16 + col]
                                             : myLds[256 + m * 16 + (col - 16)];
                outRow[obase + dx * HW_] = val;       // p = dyi*9+dx strides by H*W
            }
        }
        __syncthreads();
        off0 += PW_;                                  // next dy row
    }
}

// ---------------- fallback (no scratch): self-contained masked kernel -----------
__global__ __launch_bounds__(192) void cost_volume_wmma_direct(
    const float* __restrict__ f1, const float* __restrict__ f2,
    float* __restrict__ out)
{
    __shared__ float lds[6 * 512];
    const int b  = blockIdx.x / H_;
    const int yy = blockIdx.x % H_;
    const int tid  = threadIdx.x;
    const int lane = tid & 31;
    const int wave = tid >> 5;
    const int x0   = wave * 16;
    const int n    = lane & 15;
    const int half = lane >> 4;

    const float* f1row  = f1 + b * CHW_ + yy * W_;
    const float* f2base = f2 + b * CHW_;
    float*       outRow = out + b * (P_ * P_) * HW_ + yy * W_;

    Frag A[8];
    const int aoff = x0 + n;
    #pragma unroll
    for (int kc = 0; kc < 8; ++kc) {
        const float* f1kc = f1row + kc * 32 * HW_;
        #pragma unroll
        for (int v = 0; v < 8; ++v) {
            const int k0 = (half << 4) + (v << 1);
            A[kc].u[v] = packbf(f1kc[aoff + k0 * HW_], f1kc[aoff + (k0 + 1) * HW_]);
        }
    }

    float* myLds = lds + wave * 512;
    for (int dyi = 0; dyi < P_; ++dyi) {
        const int  y2  = yy + dyi - 4;
        const bool yok = (y2 >= 0) && (y2 < H_);
        const int  y2c = (y2 < 0) ? 0 : ((y2 >= H_) ? (H_ - 1) : y2);
        const int  px0 = x0 - 4 + n;
        const int  px1 = x0 + 12 + n;
        const bool ok0 = yok && (px0 >= 0);
        const bool ok1 = yok && (px1 < W_);
        const int  px0c = (px0 < 0) ? 0 : px0;
        const int  px1c = (px1 >= W_) ? (W_ - 1) : px1;
        const unsigned m0 = ok0 ? 0xFFFFFFFFu : 0u;
        const unsigned m1 = ok1 ? 0xFFFFFFFFu : 0u;
        const int off0 = y2c * W_ + px0c;
        const int off1 = y2c * W_ + px1c;

        v8f c0 = {}; v8f c1 = {};
        #pragma unroll
        for (int kc = 0; kc < 8; ++kc) {
            const float* f2kc = f2base + kc * 32 * HW_;
            Frag Bf0, Bf1;
            #pragma unroll
            for (int v = 0; v < 8; ++v) {
                const int k0 = (half << 4) + (v << 1);
                Bf0.u[v] = packbf(f2kc[off0 + k0 * HW_], f2kc[off0 + (k0 + 1) * HW_]) & m0;
                Bf1.u[v] = packbf(f2kc[off1 + k0 * HW_], f2kc[off1 + (k0 + 1) * HW_]) & m1;
            }
            c0 = __builtin_amdgcn_wmma_f32_16x16x32_bf16(
                     false, A[kc].v, false, Bf0.v, (short)0, c0, false, false);
            c1 = __builtin_amdgcn_wmma_f32_16x16x32_bf16(
                     false, A[kc].v, false, Bf1.v, (short)0, c1, false, false);
        }

        #pragma unroll
        for (int r = 0; r < 8; ++r) {
            myLds[((half << 3) + r) * 16 + n]       = c0[r];
            myLds[256 + ((half << 3) + r) * 16 + n] = c1[r];
        }
        __syncthreads();
        if (lane < 16) {
            const int m = lane;
            const int obase = (dyi * P_) * HW_ + x0 + m;
            #pragma unroll
            for (int dx = 0; dx < P_; ++dx) {
                const int col = m + dx;
                const float val = (col < 16) ? myLds[m * 16 + col]
                                             : myLds[256 + m * 16 + (col - 16)];
                outRow[obase + dx * HW_] = val;
            }
        }
        __syncthreads();
    }
}

extern "C" void kernel_launch(void* const* d_in, const int* in_sizes, int n_in,
                              void* d_out, int out_size, void* d_ws, size_t ws_size,
                              hipStream_t stream) {
    const float* feat1 = (const float*)d_in[0];
    const float* feat2 = (const float*)d_in[1];
    float* out = (float*)d_out;
    (void)in_sizes; (void)n_in; (void)out_size;

    const size_t need = 2ull * NPADDED_ * sizeof(unsigned);   // ~47.7 MB
    if (ws_size >= need && d_ws != nullptr) {
        unsigned* w1 = (unsigned*)d_ws;
        unsigned* w2 = w1 + NPADDED_;
        const int pb = (NPADDED_ + 255) / 256;                // 23296 blocks
        pack_pairs_padded<<<pb, 256, 0, stream>>>(feat1, w1);
        pack_pairs_padded<<<pb, 256, 0, stream>>>(feat2, w2);
        cost_volume_wmma_padded<<<dim3(B_ * H_), dim3(192), 0, stream>>>(w1, w2, out);
    } else {
        cost_volume_wmma_direct<<<dim3(B_ * H_), dim3(192), 0, stream>>>(feat1, feat2, out);
    }
}
